// ClusterNet_75642964017166
// MI455X (gfx1250) — matrix operand order repaired and checked by reading, not compiled, not run on hardware
//
#include <hip/hip_runtime.h>
#include <math.h>

typedef __attribute__((ext_vector_type(16))) __bf16 v16bf;
typedef __attribute__((ext_vector_type(8)))  float  v8f;

#define Bn 32
#define Nn 4096
#define Dn 256
#define Kn 64
#define INV_EPS 20.0f   // 1/0.05

__device__ __forceinline__ float wave_red_sum(float v) {
#pragma unroll
  for (int m = 16; m >= 1; m >>= 1) v += __shfl_xor(v, m, 32);
  return v;
}

// ---------------------------------------------------------------------------
// L2-normalize rows of a [rows, 256] f32 matrix, emit bf16. One wave per row.
// ---------------------------------------------------------------------------
__global__ void norm_rows_to_bf16(const float* __restrict__ src,
                                  __bf16* __restrict__ dst) {
  int wave = threadIdx.x >> 5, lane = threadIdx.x & 31;
  int row  = blockIdx.x * 8 + wave;
  const float* p = src + (size_t)row * Dn + lane * 8;
  float v[8]; float ss = 0.f;
#pragma unroll
  for (int j = 0; j < 8; ++j) { v[j] = p[j]; ss += v[j] * v[j]; }
  ss = wave_red_sum(ss);
  float sc = 1.0f / fmaxf(sqrtf(ss), 1e-12f);
  __bf16* q = dst + (size_t)row * Dn + lane * 8;
#pragma unroll
  for (int j = 0; j < 8; ++j) q[j] = (__bf16)(v[j] * sc);
}

// ---------------------------------------------------------------------------
// Per-row inverse L2 norm of z. One wave per row.
// ---------------------------------------------------------------------------
__global__ void z_inv_norm(const float* __restrict__ z,
                           float* __restrict__ inv_nz) {
  int wave = threadIdx.x >> 5, lane = threadIdx.x & 31;
  int row  = blockIdx.x * 8 + wave;
  const float* p = z + (size_t)row * Dn + lane * 8;
  float ss = 0.f;
#pragma unroll
  for (int j = 0; j < 8; ++j) { float t = p[j]; ss += t * t; }
  ss = wave_red_sum(ss);
  if (lane == 0) inv_nz[row] = 1.0f / fmaxf(sqrtf(ss), 1e-12f);
}

// ---------------------------------------------------------------------------
// s = z_n @ P^T via v_wmma_f32_16x16x32_bf16, fused softmax(k) + exp(./eps).
// Wave computes a 16(n) x 64(k) strip; block = 8 waves = 128 rows.
// Grid: (N/128, B). Writes Q0[b, n, k].
//
// A operand (16-bit A 16x32, contiguous K): lane L (L<16) row M=L holds
//   K {0..15} in v0..v7; lanes 16..31 hold K {16..31}
//   for the same rows -> two 8-float runs at +kb and +kb+8.
// B operand (16-bit B 32x16): lane L holds column N = L&15 with the
//   contiguous K range hi*16 .. hi*16+15 across v0..v7 -> one 32B chunk.
// ---------------------------------------------------------------------------
__global__ void gemm_softmax_exp(const float* __restrict__ z,
                                 const float* __restrict__ inv_nz,
                                 const __bf16* __restrict__ protosb,
                                 float* __restrict__ Q0) {
  int b    = blockIdx.y;
  int wave = threadIdx.x >> 5, lane = threadIdx.x & 31;
  int lane15 = lane & 15;
  int hi     = lane >> 4;      // 0 or 1
  int kb     = hi * 16;
  int rowBase = blockIdx.x * 128 + wave * 16;
  int arow    = rowBase + lane15;

  const float* zrow = z + ((size_t)b * Nn + arow) * Dn;
  float sc = inv_nz[b * Nn + arow];
  const __bf16* pb = protosb + (size_t)b * Kn * Dn;

  v8f acc[4] = {};   // 4 column tiles of 16 -> k = 0..63
#pragma unroll
  for (int kc = 0; kc < 8; ++kc) {       // K = 256 in chunks of 32
    int c0 = kc * 32;
    // A fragment: scale z row by inv-norm, convert to bf16
    v16bf a;
    const float* p0 = zrow + c0 + kb;
    const float* p1 = zrow + c0 + kb + 8;
#pragma unroll
    for (int j = 0; j < 8; ++j) {
      a[j]     = (__bf16)(p0[j] * sc);
      a[8 + j] = (__bf16)(p1[j] * sc);
    }
#pragma unroll
    for (int ct = 0; ct < 4; ++ct) {
      int col = ct * 16 + lane15;        // prototype index (column of B)
      // B fragment: contiguous K = c0 + hi*16 .. +15 of prototype row `col`
      const __bf16* pr = pb + (size_t)col * Dn + c0 + hi * 16;
      v16bf bm = *(const v16bf*)pr;      // 32B aligned contiguous load
      acc[ct] = __builtin_amdgcn_wmma_f32_16x16x32_bf16(
          false, a, false, bm, (short)0, acc[ct], false, false);
    }
  }

  // C/D layout: vgpr i, lane L: row = rowBase + i + 8*(L>=16), col = ct*16 + (L&15)
  // Softmax over the 64 k-values of each row (spread over 16 lanes x 4 tiles).
#pragma unroll
  for (int i = 0; i < 8; ++i) {
    int rr = rowBase + i + hi * 8;
    float v0 = acc[0][i], v1 = acc[1][i], v2 = acc[2][i], v3 = acc[3][i];
    float m = fmaxf(fmaxf(v0, v1), fmaxf(v2, v3));
#pragma unroll
    for (int s = 8; s >= 1; s >>= 1) m = fmaxf(m, __shfl_xor(m, s, 16));
    float e0 = __expf(v0 - m), e1 = __expf(v1 - m);
    float e2 = __expf(v2 - m), e3 = __expf(v3 - m);
    float sum = e0 + e1 + e2 + e3;
#pragma unroll
    for (int s = 8; s >= 1; s >>= 1) sum += __shfl_xor(sum, s, 16);
    float inv = 1.0f / sum;
    float* qrow = Q0 + ((size_t)b * Nn + rr) * Kn;
    qrow[0 * 16 + lane15] = __expf(e0 * inv * INV_EPS);
    qrow[1 * 16 + lane15] = __expf(e1 * inv * INV_EPS);
    qrow[2 * 16 + lane15] = __expf(e2 * inv * INV_EPS);
    qrow[3 * 16 + lane15] = __expf(e3 * inv * INV_EPS);
  }
}

// ---------------------------------------------------------------------------
// Sinkhorn via scaling factors: Q = Q0 * r[k] * c[n].  The initial global
// normalization cancels against the first row-normalize, so:
//   r[k] = 1/(K * sum_n Q0[n,k]*c[n]),  c[n] = 1/(N * sum_k Q0[n,k]*r[k])
// One 1024-thread workgroup per batch; c lives in LDS.
// ---------------------------------------------------------------------------
__global__ void sinkhorn_kernel(const float* __restrict__ Q0,
                                float* __restrict__ rbuf,
                                float* __restrict__ cbuf) {
  __shared__ float c_l[Nn];
  __shared__ float r_l[Kn];
  __shared__ float red[16 * Kn];
  int b = blockIdx.x, t = threadIdx.x;
  const float* Qb = Q0 + (size_t)b * Nn * Kn;

  for (int j = t; j < Nn; j += 1024) c_l[j] = 1.0f;
  __syncthreads();

  for (int it = 0; it < 3; ++it) {
    // r update: per-k sum over n (threads = 64 k x 16 n-groups)
    int k = t & 63, g = t >> 6;
    float s = 0.f;
    for (int j = g; j < Nn; j += 16) s += Qb[(size_t)j * Kn + k] * c_l[j];
    red[g * Kn + k] = s;
    __syncthreads();
    if (t < Kn) {
      float a = 0.f;
#pragma unroll
      for (int g2 = 0; g2 < 16; ++g2) a += red[g2 * Kn + t];
      r_l[t] = 1.0f / (64.0f * a);
    }
    __syncthreads();
    // c update: per-n dot with r
    for (int j = t; j < Nn; j += 1024) {
      const float* q = Qb + (size_t)j * Kn;
      float s2 = 0.f;
#pragma unroll
      for (int kk = 0; kk < Kn; ++kk) s2 += q[kk] * r_l[kk];
      c_l[j] = 1.0f / (4096.0f * s2);
    }
    __syncthreads();
  }
  if (t < Kn) rbuf[b * Kn + t] = r_l[t];
  for (int j = t; j < Nn; j += 1024) cbuf[b * Nn + j] = c_l[j];
}

// ---------------------------------------------------------------------------
// logits[n,k] = Q0[n,k]*r[k]*c[n]*N ; argmax -> indexes. One thread per row.
// ---------------------------------------------------------------------------
__global__ void logits_argmax(const float* __restrict__ Q0,
                              const float* __restrict__ rbuf,
                              const float* __restrict__ cbuf,
                              float* __restrict__ logits,
                              float* __restrict__ idxf,
                              int* __restrict__ idxi) {
  __shared__ float r_l[Kn];
  int b = blockIdx.y;
  int row = blockIdx.x * 256 + threadIdx.x;
  if (threadIdx.x < Kn) r_l[threadIdx.x] = rbuf[b * Kn + threadIdx.x];
  __syncthreads();
  const float* q = Q0 + ((size_t)b * Nn + row) * Kn;
  float cn = cbuf[b * Nn + row] * 4096.0f;
  float* lrow = logits + ((size_t)b * Nn + row) * Kn;
  float best = -1e30f; int bi = 0;
#pragma unroll
  for (int kk = 0; kk < Kn; ++kk) {
    float v = q[kk] * r_l[kk] * cn;
    lrow[kk] = v;
    if (v > best) { best = v; bi = kk; }
  }
  idxf[b * Nn + row] = (float)bi;
  idxi[b * Nn + row] = bi;
}

// ---------------------------------------------------------------------------
// prototypes accumulation: accum[b,k,d] = sum_n onehot[n,k]*z[n,d].
// Grid (D/64, B): each block owns a disjoint (batch, d-slice) -> plain
// global stores; contention handled with LDS ds_add_f32 atomics.
// ---------------------------------------------------------------------------
__global__ void proto_accum(const float* __restrict__ z,
                            const int* __restrict__ idxi,
                            float* __restrict__ accum,
                            float* __restrict__ counts) {
  __shared__ float acc_l[Kn * 64];
  __shared__ float cnt_l[Kn];
  int b = blockIdx.y;
  int ds = blockIdx.x * 64;
  int t = threadIdx.x, dcol = t & 63, g = t >> 6;  // g in 0..3
  for (int j = t; j < Kn * 64; j += 256) acc_l[j] = 0.f;
  if (blockIdx.x == 0 && t < Kn) cnt_l[t] = 0.f;
  __syncthreads();
  for (int j = g; j < Nn; j += 4) {
    int kk = idxi[b * Nn + j];
    float v = z[((size_t)b * Nn + j) * Dn + ds + dcol];
    atomicAdd(&acc_l[kk * 64 + dcol], v);
    if (blockIdx.x == 0 && dcol == 0) atomicAdd(&cnt_l[kk], 1.0f);
  }
  __syncthreads();
  for (int kk = g; kk < Kn; kk += 4)
    accum[((size_t)b * Kn + kk) * Dn + ds + dcol] = acc_l[kk * 64 + dcol];
  if (blockIdx.x == 0 && t < Kn) counts[b * Kn + t] = cnt_l[t];
}

// ---------------------------------------------------------------------------
// prototypes = l2_normalize(nan_to_num(accum/count)) -> bf16. Wave per row.
// ---------------------------------------------------------------------------
__global__ void finalize_protos(const float* __restrict__ accum,
                                const float* __restrict__ counts,
                                __bf16* __restrict__ dst) {
  int wave = threadIdx.x >> 5, lane = threadIdx.x & 31;
  int row  = blockIdx.x * 8 + wave;   // row in [0, B*K)
  float cnt = counts[row];
  float ic  = cnt > 0.f ? 1.0f / cnt : 0.f;  // count==0 => row of zeros (nan_to_num)
  const float* p = accum + (size_t)row * Dn + lane * 8;
  float v[8]; float ss = 0.f;
#pragma unroll
  for (int j = 0; j < 8; ++j) { v[j] = p[j] * ic; ss += v[j] * v[j]; }
  ss = wave_red_sum(ss);
  float sc = 1.0f / fmaxf(sqrtf(ss), 1e-12f);
  __bf16* q = dst + (size_t)row * Dn + lane * 8;
#pragma unroll
  for (int j = 0; j < 8; ++j) q[j] = (__bf16)(v[j] * sc);
}

// ---------------------------------------------------------------------------
extern "C" void kernel_launch(void* const* d_in, const int* in_sizes, int n_in,
                              void* d_out, int out_size, void* d_ws, size_t ws_size,
                              hipStream_t stream) {
  (void)in_sizes; (void)n_in; (void)out_size; (void)ws_size;
  const float* z          = (const float*)d_in[0];
  const float* proto_init = (const float*)d_in[1];

  float* logits = (float*)d_out;                    // [B,N,K]
  float* idxf   = logits + (size_t)Bn * Nn * Kn;    // [B,N] as float

  char* w = (char*)d_ws;
  auto alloc = [&](size_t bytes) -> char* {
    char* p = w; w += (bytes + 255) & ~(size_t)255; return p;
  };
  float*  inv_nz  = (float*)alloc((size_t)Bn * Nn * 4);
  float*  Q0      = (float*)alloc((size_t)Bn * Nn * Kn * 4);
  float*  rbuf    = (float*)alloc((size_t)Bn * Kn * 4);
  float*  cbuf    = (float*)alloc((size_t)Bn * Nn * 4);
  float*  accum   = (float*)alloc((size_t)Bn * Kn * Dn * 4);
  float*  counts  = (float*)alloc((size_t)Bn * Kn * 4);
  int*    idxi    = (int*)alloc((size_t)Bn * Nn * 4);
  __bf16* protosb = (__bf16*)alloc((size_t)Bn * Kn * Dn * 2);

  norm_rows_to_bf16<<<Bn * Kn / 8, 256, 0, stream>>>(proto_init, protosb);
  z_inv_norm<<<Bn * Nn / 8, 256, 0, stream>>>(z, inv_nz);

  for (int it = 0; it < 3; ++it) {
    gemm_softmax_exp<<<dim3(Nn / 128, Bn), 256, 0, stream>>>(z, inv_nz, protosb, Q0);
    sinkhorn_kernel<<<Bn, 1024, 0, stream>>>(Q0, rbuf, cbuf);
    logits_argmax<<<dim3(Nn / 256, Bn), 256, 0, stream>>>(Q0, rbuf, cbuf,
                                                          logits, idxf, idxi);
    if (it < 2) {
      proto_accum<<<dim3(Dn / 64, Bn), 256, 0, stream>>>(z, idxi, accum, counts);
      finalize_protos<<<Bn * Kn / 8, 256, 0, stream>>>(accum, counts, protosb);
    }
  }
}
